// TopKPerceptronRouter_44401371906542
// MI455X (gfx1250) — matrix operand order repaired and checked by reading, not compiled, not run on hardware
//
#include <hip/hip_runtime.h>
#include <hip/hip_bf16.h>
#include <math.h>

// ---- problem constants (from reference) ----
#define D_DIM     4096
#define NEXP      64
#define T_TOK     (4 * 4096)       // B*S tokens

// ---- tiling ----
#define KB        128              // d-chunk staged in LDS per outer iteration
#define NBLK      (D_DIM / KB)     // 32
#define KSTEPS    (KB / 32)        // 4 WMMA k-steps per chunk
#define WPB       4                // waves per workgroup
#define TOK_WAVE  16
#define TOK_WG    (WPB * TOK_WAVE) // 64 tokens per workgroup
#define PITCHB    144              // bf16 elems per W row in LDS (128 + 16 pad, keeps 16B alignment)
#define PITCHB_Q  (PITCHB / 4)     // row pitch in 8-byte units (= 36)
#define PITCHL    65               // floats per logit row in LDS (64 + 1 pad)

#define USE_ASYNC_STAGE 1          // gfx1250 global_load_async_to_lds_b128 staging

typedef __attribute__((ext_vector_type(16))) __bf16        v16bf;
typedef __attribute__((ext_vector_type(8)))  float         v8f;
typedef __attribute__((ext_vector_type(4)))  float         v4f;
typedef __attribute__((ext_vector_type(2)))  unsigned int  u32x2;
typedef __attribute__((ext_vector_type(4)))  unsigned int  u32x4;

static __device__ __forceinline__ unsigned pack_bf16(float a, float b) {
    unsigned short lo = __builtin_bit_cast(unsigned short, (__bf16)a);
    unsigned short hi = __builtin_bit_cast(unsigned short, (__bf16)b);
    return (unsigned)lo | ((unsigned)hi << 16);
}

#if USE_ASYNC_STAGE
// Copy 16 bytes global -> LDS via the async tensor/LDS DMA path (ASYNCcnt).
static __device__ __forceinline__ void cp_b128_async(unsigned lds_byte_off, const void* gsrc) {
    asm volatile("global_load_async_to_lds_b128 %0, %1, off"
                 :: "v"(lds_byte_off),
                    "v"((unsigned long long)(__SIZE_TYPE__)gsrc)
                 : "memory");
}
static __device__ __forceinline__ void wait_async_done() {
    asm volatile("s_wait_asynccnt 0x0" ::: "memory");
}
#endif

// ---- pre-kernel: W fp32 (64x4096) -> packed bf16 pairs in d_ws ----
__global__ __launch_bounds__(256)
void convert_W_bf16(const float* __restrict__ W, unsigned* __restrict__ Wb) {
    const int i = blockIdx.x * 256 + threadIdx.x;     // 0 .. 64*4096/2 - 1
    Wb[i] = pack_bf16(W[2 * i + 0], W[2 * i + 1]);
}

__global__ __launch_bounds__(128)
void router_gemm_topk(const float* __restrict__ x,
                      const __bf16* __restrict__ Wb,     // bf16 W, row-major 64x4096
                      const float* __restrict__ bias_g,
                      int*   __restrict__ idx_out,
                      float* __restrict__ w_out)
{
    __shared__ __bf16 Wbuf[2][NEXP * PITCHB];   // 2 x 18432 B double-buffered W chunk
    __shared__ float  Llds[TOK_WG * PITCHL];    // 16640 B logits for top-k

    const int tid  = threadIdx.x;
    const int wave = tid >> 5;
    const int lane = tid & 31;
    const int col  = lane & 15;      // expert column / token row within tile
    const int hi   = lane >> 4;      // lane half (selects K sub-ranges)

    const int tokBaseWG = blockIdx.x * TOK_WG;
    const int tok       = tokBaseWG + wave * TOK_WAVE + col;  // this lane's A row
    const float* __restrict__ xrow = x + (size_t)tok * D_DIM;

    // staging assignment: 64 rows x 16 b128-units = 1024 units / 128 threads = 8 each
    const int su_e = (tid * 8) >> 4;          // 8 consecutive units stay within a row? no ->
    (void)su_e;

    float bv[4];
#pragma unroll
    for (int t = 0; t < 4; ++t) bv[t] = bias_g[t * 16 + col];

    v8f acc[4];
#pragma unroll
    for (int t = 0; t < 4; ++t) acc[t] = (v8f){0.f,0.f,0.f,0.f,0.f,0.f,0.f,0.f};

    // ---- stage one 64 x 128 bf16 chunk of W into LDS buffer p ----
    auto stage = [&](int p, int kbase) {
#pragma unroll
        for (int i = 0; i < 8; ++i) {
            const int u = tid + i * 128;      // 0..1023 b128-units
            const int e = u >> 4;             // expert row 0..63
            const int c = u & 15;             // 16B unit within row
            const __bf16* gsrc = Wb + (size_t)e * D_DIM + kbase + c * 8;
#if USE_ASYNC_STAGE
            const unsigned loff =
                (unsigned)(__SIZE_TYPE__)&Wbuf[p][0] + (unsigned)(e * (PITCHB * 2) + c * 16);
            cp_b128_async(loff, gsrc);
#else
            u32x4 v = *(const u32x4*)gsrc;
            *(u32x4*)((char*)&Wbuf[p][0] + e * (PITCHB * 2) + c * 16) = v;
#endif
        }
    };

    // prologue: stage chunk 0
    stage(0, 0);
#if USE_ASYNC_STAGE
    wait_async_done();
#endif
    __syncthreads();

#pragma unroll 1
    for (int blk = 0; blk < NBLK; ++blk) {
        const int p     = blk & 1;
        const int kbase = blk * KB;

        // issue staging of next chunk first so it overlaps compute
        if (blk + 1 < NBLK) stage(p ^ 1, kbase + KB);

        const u32x2* __restrict__ Wq = (const u32x2*)&Wbuf[p][0];

#pragma unroll
        for (int ks = 0; ks < KSTEPS; ++ks) {
            // ---- A fragment (contiguous k = 16h + s order, matches B) ----
            // lanes 0-15: K = 0..15 ; lanes 16-31: K = 16..31
            const int k0 = kbase + ks * 32 + hi * 16;
            v4f a0 = *(const v4f*)(xrow + k0);
            v4f a1 = *(const v4f*)(xrow + k0 + 4);
            v4f a2 = *(const v4f*)(xrow + k0 + 8);
            v4f a3 = *(const v4f*)(xrow + k0 + 12);
            v16bf va;
#pragma unroll
            for (int i = 0; i < 4; ++i) {
                va[i]      = (__bf16)a0[i];
                va[4 + i]  = (__bf16)a1[i];
                va[8 + i]  = (__bf16)a2[i];
                va[12 + i] = (__bf16)a3[i];
            }

            // ---- all 4 B fragments first (batch the ds_loads) ----
            // lanes 0-15: K = 0..15 ; lanes 16-31: K = 16..31
            const int kkq = (ks * 32 + hi * 16) >> 2;   // u32x2 index within row
            union BU { v16bf bf; u32x2 q[4]; } ub[4];
#pragma unroll
            for (int t = 0; t < 4; ++t) {
                const u32x2* rp = Wq + (t * 16 + col) * PITCHB_Q + kkq;
                ub[t].q[0] = rp[0];
                ub[t].q[1] = rp[1];
                ub[t].q[2] = rp[2];
                ub[t].q[3] = rp[3];
            }
            // ---- then 4 back-to-back WMMAs ----
#pragma unroll
            for (int t = 0; t < 4; ++t) {
                acc[t] = __builtin_amdgcn_wmma_f32_16x16x32_bf16(
                    false, va, false, ub[t].bf, (short)0, acc[t], false, false);
            }
        }

#if USE_ASYNC_STAGE
        wait_async_done();     // staged chunk blk+1 fully landed in LDS
#endif
        __syncthreads();       // all waves done reading buf p; buf p^1 visible
    }

    // ---- scatter C tiles (+bias) to LDS logits [token][expert] ----
#pragma unroll
    for (int t = 0; t < 4; ++t) {
#pragma unroll
        for (int j = 0; j < 8; ++j) {
            const int row = wave * TOK_WAVE + hi * 8 + j;  // token within WG
            const int e   = t * 16 + col;                  // expert
            Llds[row * PITCHL + e] = acc[t][j] + bv[t];
        }
    }
    __syncthreads();

    // ---- per-token top-2 + 2-way softmax ----
    if (tid < TOK_WG) {
        const float* lrow = Llds + tid * PITCHL;
        float m1 = -INFINITY, m2 = -INFINITY;
        int   i1 = 0, i2 = 0;
#pragma unroll 4
        for (int e = 0; e < NEXP; ++e) {
            float v = lrow[e];
            if (v > m1)      { m2 = m1; i2 = i1; m1 = v; i1 = e; }
            else if (v > m2) { m2 = v; i2 = e; }
        }
        const float e2  = __expf(m2 - m1);
        const float inv = 1.0f / (1.0f + e2);
        const size_t t  = (size_t)(tokBaseWG + tid);
        idx_out[t * 2 + 0] = i1;
        idx_out[t * 2 + 1] = i2;
        w_out [t * 2 + 0] = inv;
        w_out [t * 2 + 1] = e2 * inv;
    }
}

extern "C" void kernel_launch(void* const* d_in, const int* in_sizes, int n_in,
                              void* d_out, int out_size, void* d_ws, size_t ws_size,
                              hipStream_t stream) {
    const float* x = (const float*)d_in[0];   // (4, 4096, 4096) fp32
    const float* W = (const float*)d_in[1];   // (64, 4096) fp32
    const float* b = (const float*)d_in[2];   // (64,) fp32

    // scratch: bf16 copy of W (64*4096*2 B = 512 KiB)
    unsigned* Wb_u32 = (unsigned*)d_ws;

    convert_W_bf16<<<dim3((NEXP * D_DIM / 2) / 256), dim3(256), 0, stream>>>(W, Wb_u32);

    int*   idx_out = (int*)d_out;
    float* w_out   = (float*)d_out + (size_t)T_TOK * 2;

    router_gemm_topk<<<dim3(T_TOK / TOK_WG), dim3(128), 0, stream>>>(
        x, (const __bf16*)Wb_u32, b, idx_out, w_out);
}
